// GhostMLPModel_78271484003126
// MI455X (gfx1250) — compile-verified, matrix operand order checked by reading, not run on hardware
//
#include <hip/hip_runtime.h>

typedef __attribute__((ext_vector_type(16))) _Float16 v16h;
typedef __attribute__((ext_vector_type(8)))  float    v8f;
typedef __attribute__((ext_vector_type(4)))  int      i32x4;
typedef _Float16 h16;

union V16U { v16h v; ulonglong2 q[2]; };

#ifndef __has_builtin
#define __has_builtin(x) 0
#endif
#if __has_builtin(__builtin_amdgcn_global_load_async_to_lds_b128) && \
    __has_builtin(__builtin_amdgcn_s_wait_asynccnt)
#define HAVE_ASYNC_LDS 1
#else
#define HAVE_ASYNC_LDS 0
#endif

#define AS1 __attribute__((address_space(1)))
#define AS3 __attribute__((address_space(3)))

// ---------------------------------------------------------------------------
// Farthest point sampling: one block per batch. Strided point access so the
// same kernel handles (B,3,N) channel-major and (B,M,3) row-major layouts.
// ---------------------------------------------------------------------------
__global__ void __launch_bounds__(256)
fps_kernel(const float* __restrict__ pts, int n, int npoint,
           long sb, long sn, long sc,
           int* __restrict__ idx_out, float* __restrict__ new_xyz) {
  __shared__ float dist[4096];
  __shared__ float redv[256];
  __shared__ int   redi[256];
  __shared__ int   far_s;
  const int b = blockIdx.x;
  const int tid = threadIdx.x;
  const float* base = pts + (long)b * sb;
  for (int i = tid; i < n; i += 256) dist[i] = 1e10f;
  if (tid == 0) far_s = 0;
  __syncthreads();
  for (int it = 0; it < npoint; ++it) {
    const int far = far_s;
    const float fx = base[(long)far * sn + 0 * sc];
    const float fy = base[(long)far * sn + 1 * sc];
    const float fz = base[(long)far * sn + 2 * sc];
    if (tid == 0) {
      idx_out[b * npoint + it] = far;
      new_xyz[((long)b * npoint + it) * 3 + 0] = fx;
      new_xyz[((long)b * npoint + it) * 3 + 1] = fy;
      new_xyz[((long)b * npoint + it) * 3 + 2] = fz;
    }
    float best = -1.0f; int bi = 0;
    for (int i = tid; i < n; i += 256) {
      const float dx = base[(long)i * sn + 0 * sc] - fx;
      const float dy = base[(long)i * sn + 1 * sc] - fy;
      const float dz = base[(long)i * sn + 2 * sc] - fz;
      float d = dx * dx + dy * dy + dz * dz;
      const float o = dist[i];
      d = d < o ? d : o;
      dist[i] = d;
      if (d > best) { best = d; bi = i; }
    }
    redv[tid] = best; redi[tid] = bi;
    __syncthreads();
    for (int off = 128; off > 0; off >>= 1) {
      if (tid < off) {
        const float v2 = redv[tid + off];
        if (v2 > redv[tid] || (v2 == redv[tid] && redi[tid + off] < redi[tid])) {
          redv[tid] = v2; redi[tid] = redi[tid + off];
        }
      }
      __syncthreads();
    }
    if (tid == 0) far_s = redi[0];
    __syncthreads();
  }
}

// ---------------------------------------------------------------------------
// Ball query + gather, SA1 (input channels = 3 rel-xyz, padded to 32 f16).
// One wave per query point; in-order first-K selection via ballot prefix.
// ---------------------------------------------------------------------------
__global__ void __launch_bounds__(32)
bq_sa1(const float* __restrict__ xyz, const float* __restrict__ new_xyz,
       float r2, int K, h16* __restrict__ Xb) {
  const int q = blockIdx.x;              // b*512 + s
  const int b = q >> 9;
  const int lane = threadIdx.x;
  const float* xb = xyz + (size_t)b * 3 * 4096;
  const float cx = new_xyz[q * 3 + 0];
  const float cy = new_xyz[q * 3 + 1];
  const float cz = new_xyz[q * 3 + 2];
  __shared__ float fst[3];
  int cnt = 0;
  for (int base = 0; base < 4096 && cnt < K; base += 32) {
    const int n = base + lane;
    const float dx = xb[n] - cx;
    const float dy = xb[n + 4096] - cy;
    const float dz = xb[n + 8192] - cz;
    const float d2 = dx * dx + dy * dy + dz * dz;
    const bool ok = d2 <= r2;
    const unsigned mask = (unsigned)__ballot(ok ? 1 : 0);
    const int ord = __popc(mask & ((1u << lane) - 1u));
    const int slot = cnt + ord;
    if (ok && slot < K) {
      alignas(16) h16 h[32];
      h[0] = (h16)dx; h[1] = (h16)dy; h[2] = (h16)dz;
#pragma unroll
      for (int c = 3; c < 32; ++c) h[c] = (h16)0.0f;
      h16* dst = Xb + ((size_t)q * K + slot) * 32;
#pragma unroll
      for (int v = 0; v < 4; ++v)
        *(ulonglong2*)(dst + v * 8) = *(const ulonglong2*)(h + v * 8);
      if (slot == 0) { fst[0] = dx; fst[1] = dy; fst[2] = dz; }
    }
    cnt += __popc(mask);
    if (cnt > K) cnt = K;
  }
  __syncthreads();
  for (int t = lane; t < (K - cnt) * 32; t += 32) {  // pad with first hit
    const int slot = cnt + t / 32;
    const int c = t % 32;
    Xb[((size_t)q * K + slot) * 32 + c] = (h16)((c < 3) ? fst[c] : 0.0f);
  }
}

// ---------------------------------------------------------------------------
// Ball query + gather, SA2: 320 feature channels + 3 rel-xyz, padded to 352.
// ---------------------------------------------------------------------------
__global__ void __launch_bounds__(32)
bq_sa2(const float* __restrict__ l1_xyz, const float* __restrict__ l1_pts,
       const float* __restrict__ new_xyz, float r2, int K, h16* __restrict__ Xb) {
  const int q = blockIdx.x;              // b*128 + s
  const int b = q >> 7;
  const int lane = threadIdx.x;
  const float cx = new_xyz[q * 3 + 0];
  const float cy = new_xyz[q * 3 + 1];
  const float cz = new_xyz[q * 3 + 2];
  __shared__ int   fstn;
  __shared__ float fstd[3];
  int cnt = 0;
  for (int base = 0; base < 512 && cnt < K; base += 32) {
    const int n = base + lane;
    const long pr = (long)b * 512 + n;
    const float dx = l1_xyz[pr * 3 + 0] - cx;
    const float dy = l1_xyz[pr * 3 + 1] - cy;
    const float dz = l1_xyz[pr * 3 + 2] - cz;
    const float d2 = dx * dx + dy * dy + dz * dz;
    const bool ok = d2 <= r2;
    const unsigned mask = (unsigned)__ballot(ok ? 1 : 0);
    const int ord = __popc(mask & ((1u << lane) - 1u));
    const int slot = cnt + ord;
    if (ok && slot < K) {
      const float* src = l1_pts + pr * 320;
      h16* dst = Xb + ((size_t)q * K + slot) * 352;
      for (int c = 0; c < 320; c += 4) {
        const float4 v = *(const float4*)(src + c);
        dst[c + 0] = (h16)v.x; dst[c + 1] = (h16)v.y;
        dst[c + 2] = (h16)v.z; dst[c + 3] = (h16)v.w;
      }
      dst[320] = (h16)dx; dst[321] = (h16)dy; dst[322] = (h16)dz;
      for (int c = 323; c < 352; ++c) dst[c] = (h16)0.0f;
      if (slot == 0) { fstn = n; fstd[0] = dx; fstd[1] = dy; fstd[2] = dz; }
    }
    cnt += __popc(mask);
    if (cnt > K) cnt = K;
  }
  __syncthreads();
  for (int t = lane; t < (K - cnt) * 352; t += 32) {
    const int slot = cnt + t / 352;
    const int c = t % 352;
    float v;
    if (c < 320)      v = l1_pts[((long)b * 512 + fstn) * 320 + c];
    else if (c < 323) v = fstd[c - 320];
    else              v = 0.0f;
    Xb[((size_t)q * K + slot) * 352 + c] = (h16)v;
  }
}

// ---------------------------------------------------------------------------
// SA3 gather: rows = b*128+s ; [l2_xyz(3), l2_pts(640), pad] -> 672 f16.
// ---------------------------------------------------------------------------
__global__ void __launch_bounds__(128)
gather_sa3(const float* __restrict__ l2_xyz, const float* __restrict__ l2_pts,
           h16* __restrict__ Xb) {
  const int row = blockIdx.x;            // 0..2047
  h16* dst = Xb + (size_t)row * 672;
  for (int c = threadIdx.x; c < 672; c += 128) {
    float v = 0.0f;
    if (c < 3)        v = l2_xyz[row * 3 + c];
    else if (c < 643) v = l2_pts[(long)row * 640 + (c - 3)];
    dst[c] = (h16)v;
  }
}

// ---------------------------------------------------------------------------
// Fused multi-layer MLP (X@W^T * g + b, ReLU) with WMMA, activations in LDS,
// final layer max-pooled over groupK rows via atomicMax on non-negative floats.
// 128 threads = 4 waves, each wave owns a 16-row band; 64 rows per block.
// Input staging uses GLOBAL_LOAD_ASYNC_TO_LDS_B128 (ASYNCcnt) when available.
// ---------------------------------------------------------------------------
struct MlpArgs {
  const h16* X;
  const h16* W[3];
  const float* g[3];
  const float* b[3];
  int  cdim[4];        // cdim[0]=padded cin, cdim[l+1]=cout of layer l (mult of 32/16)
  int  nlayers;
  long rows;
  int  groupK;
  float* pool;         // [rows/groupK, pool_stride] f32, pre-zeroed
  int  pool_stride;
  int  pool_offset;
  int  buf0_elems;     // per-row f16 capacity of LDS buffer 0
};

__global__ void __launch_bounds__(128) fused_mlp_pool(MlpArgs a) {
  extern __shared__ h16 lds[];
  h16* buf0 = lds;
  h16* buf1 = lds + (size_t)64 * a.buf0_elems;
  const int tid  = threadIdx.x;
  const int wave = tid >> 5;
  const int lane = tid & 31;
  const int gh   = lane >> 4;        // half-wave group (K/N offset selector)
  const int m16  = lane & 15;
  const long row0 = (long)blockIdx.x * 64;

  { // stage 64 input rows into LDS (zero-fill past a.rows)
    const int c0  = a.cdim[0];
    const int vpr = c0 >> 3;         // 16B vectors per row
    for (int i = tid; i < 64 * vpr; i += 128) {
      const int r  = i / vpr;
      const int cc = (i - r * vpr) << 3;
      h16* ldst = buf0 + (long)r * c0 + cc;
      if (row0 + r < a.rows) {
        const h16* gsrc = a.X + (row0 + r) * (long)c0 + cc;
#if HAVE_ASYNC_LDS
        __builtin_amdgcn_global_load_async_to_lds_b128(
            (AS1 i32x4*)gsrc, (AS3 i32x4*)ldst, 0, 0);
#else
        *(ulonglong2*)ldst = *(const ulonglong2*)gsrc;
#endif
      } else {
        ulonglong2 z; z.x = 0ull; z.y = 0ull;
        *(ulonglong2*)ldst = z;
      }
    }
  }
#if HAVE_ASYNC_LDS
  __builtin_amdgcn_s_wait_asynccnt(0);
#endif
  __syncthreads();

  h16* A = buf0;
  h16* O = buf1;
  for (int l = 0; l < a.nlayers; ++l) {
    const int cin  = a.cdim[l];
    const int cout = a.cdim[l + 1];
    const h16* Wl  = a.W[l];
    const bool last = (l == a.nlayers - 1);
    const int  mrow = wave << 4;
    const h16* Arow = A + (long)(mrow + m16) * cin + 8 * gh;
    for (int nt = 0; nt < (cout >> 4); ++nt) {
      const int ch = (nt << 4) + m16;
      const h16* Wrow = Wl + (long)ch * cin + 8 * gh;
      // speculative prefetch of next n-tile's weight row (global_prefetch_b8)
      __builtin_prefetch((const void*)(Wrow + 16 * (long)cin), 0, 0);
      v8f acc = {};
      for (int k = 0; k < cin; k += 32) {
        V16U av, bv;
        av.q[0] = *(const ulonglong2*)(Arow + k);
        av.q[1] = *(const ulonglong2*)(Arow + k + 16);
        bv.q[0] = *(const ulonglong2*)(Wrow + k);
        bv.q[1] = *(const ulonglong2*)(Wrow + k + 16);
        acc = __builtin_amdgcn_wmma_f32_16x16x32_f16(
            false, av.v, false, bv.v, (short)0, acc, false, false);
      }
      const float gv = a.g[l][ch];
      const float bb = a.b[l][ch];
#pragma unroll
      for (int i = 0; i < 8; ++i) {
        float y = acc[i] * gv + bb;
        y = y > 0.0f ? y : 0.0f;
        const int rl = mrow + i + (gh << 3);
        if (!last) {
          O[(long)rl * cout + ch] = (h16)y;
        } else {
          const long grow = row0 + rl;
          if (grow < a.rows) {
            const long grp = grow / a.groupK;
            unsigned* p = (unsigned*)(a.pool + grp * (long)a.pool_stride +
                                      a.pool_offset + ch);
            atomicMax(p, __float_as_uint(y));   // monotone for y >= 0
          }
        }
      }
    }
    __syncthreads();
    h16* t = A; A = O; O = t;
  }
}

// ---------------------------------------------------------------------------
// Small utility kernels.
// ---------------------------------------------------------------------------
__global__ void fill_zero_f32(float* p, long n) {
  const long i = (long)blockIdx.x * blockDim.x + threadIdx.x;
  if (i < n) p[i] = 0.0f;
}
__global__ void copy_f32(const float* __restrict__ s, float* __restrict__ d, long n) {
  const long i = (long)blockIdx.x * blockDim.x + threadIdx.x;
  if (i < n) d[i] = s[i];
}
// f32 [rows,cin] -> f16 [rows,cpad] zero-padded (used for weights & activations)
__global__ void pack_rows(const float* __restrict__ in, h16* __restrict__ out,
                          int rows, int cin, int cpad) {
  const long i = (long)blockIdx.x * blockDim.x + threadIdx.x;
  const long total = (long)rows * cpad;
  if (i >= total) return;
  const int r = (int)(i / cpad);
  const int c = (int)(i - (long)r * cpad);
  out[i] = (h16)((c < cin) ? in[(long)r * cin + c] : 0.0f);
}
// Ghost cheap branch: x2 = relu(x1 * Wc[:,0] * gc + bc); f = [x1, x2]
__global__ void ghost_cheap(const float* __restrict__ x1, const float* __restrict__ Wc,
                            const float* __restrict__ gc, const float* __restrict__ bc,
                            float* __restrict__ f) {
  const int i = blockIdx.x * 256 + threadIdx.x;      // 16*256
  if (i >= 16 * 256) return;
  const int b = i >> 8, c = i & 255;
  const float v1 = x1[i];
  float v2 = v1 * Wc[c] * gc[c] + bc[c];
  v2 = v2 > 0.0f ? v2 : 0.0f;
  f[b * 512 + c] = v1;
  f[b * 512 + 256 + c] = v2;
}
// fc3 + log_softmax, one block per batch row.
__global__ void __launch_bounds__(64)
fc3_logsoftmax(const float* __restrict__ f, const float* __restrict__ W,
               const float* __restrict__ bias, float* __restrict__ out) {
  const int b = blockIdx.x, t = threadIdx.x;
  __shared__ float logits[40];
  __shared__ float red[64];
  if (t < 40) {
    float acc = bias[t];
    const float* fr = f + b * 512;
    const float* wr = W + t * 512;
    for (int c = 0; c < 512; c += 4) {
      const float4 fv = *(const float4*)(fr + c);
      const float4 wv = *(const float4*)(wr + c);
      acc += fv.x * wv.x + fv.y * wv.y + fv.z * wv.z + fv.w * wv.w;
    }
    logits[t] = acc;
  }
  __syncthreads();
  red[t] = (t < 40) ? logits[t] : -1e30f;
  __syncthreads();
  for (int o = 32; o > 0; o >>= 1) { if (t < o) red[t] = fmaxf(red[t], red[t + o]); __syncthreads(); }
  const float mx = red[0];
  __syncthreads();
  red[t] = (t < 40) ? __expf(logits[t] - mx) : 0.0f;
  __syncthreads();
  for (int o = 32; o > 0; o >>= 1) { if (t < o) red[t] += red[t + o]; __syncthreads(); }
  const float lse = mx + __logf(red[0]);
  if (t < 40) out[b * 40 + t] = logits[t] - lse;
}

// ---------------------------------------------------------------------------
// Host orchestration.
// ---------------------------------------------------------------------------
extern "C" void kernel_launch(void* const* d_in, const int* in_sizes, int n_in,
                              void* d_out, int out_size, void* d_ws, size_t ws_size,
                              hipStream_t stream) {
  (void)in_sizes; (void)n_in; (void)out_size; (void)ws_size;

  // ---- unpack inputs (dict order, leaves in nested order) ----
  int ix = 0;
  const float* xyz = (const float*)d_in[ix++];
  const float *W1[3][3], *G1[3][3], *B1[3][3];
  for (int br = 0; br < 3; ++br)
    for (int l = 0; l < 3; ++l) {
      W1[br][l] = (const float*)d_in[ix++];
      G1[br][l] = (const float*)d_in[ix++];
      B1[br][l] = (const float*)d_in[ix++];
    }
  const float *W2[3][3], *G2[3][3], *B2[3][3];
  for (int br = 0; br < 3; ++br)
    for (int l = 0; l < 3; ++l) {
      W2[br][l] = (const float*)d_in[ix++];
      G2[br][l] = (const float*)d_in[ix++];
      B2[br][l] = (const float*)d_in[ix++];
    }
  const float *W3[3], *G3[3], *B3[3];
  for (int l = 0; l < 3; ++l) {
    W3[l] = (const float*)d_in[ix++];
    G3[l] = (const float*)d_in[ix++];
    B3[l] = (const float*)d_in[ix++];
  }
  const float* g1W  = (const float*)d_in[ix++];
  const float* g1g  = (const float*)d_in[ix++];
  const float* g1b  = (const float*)d_in[ix++];
  const float* g1Wc = (const float*)d_in[ix++];
  const float* g1gc = (const float*)d_in[ix++];
  const float* g1bc = (const float*)d_in[ix++];
  const float* g2W  = (const float*)d_in[ix++];
  const float* g2g  = (const float*)d_in[ix++];
  const float* g2b  = (const float*)d_in[ix++];
  const float* g2Wc = (const float*)d_in[ix++];
  const float* g2gc = (const float*)d_in[ix++];
  const float* g2bc = (const float*)d_in[ix++];
  const float* fc3w = (const float*)d_in[ix++];
  const float* fc3b = (const float*)d_in[ix++];

  // ---- carve workspace (~205 MB total) ----
  char* ws = (char*)d_ws;
  size_t off = 0;
  auto alloc = [&](size_t bytes) -> void* {
    off = (off + 255) & ~(size_t)255;
    void* p = ws + off;
    off += bytes;
    return p;
  };
  float* l1_xyz = (float*)alloc(16ull * 512 * 3 * 4);
  int*   idx1   = (int*)  alloc(16ull * 512 * 4);
  float* l1_pts = (float*)alloc(16ull * 512 * 320 * 4);
  float* l2_xyz = (float*)alloc(16ull * 128 * 3 * 4);
  int*   idx2   = (int*)  alloc(16ull * 128 * 4);
  float* l2_pts = (float*)alloc(16ull * 128 * 640 * 4);
  float* l3     = (float*)alloc(16ull * 1024 * 4);
  float* x1a    = (float*)alloc(16ull * 256 * 4);
  float* f1     = (float*)alloc(16ull * 512 * 4);
  float* x1b    = (float*)alloc(16ull * 256 * 4);
  float* f2     = (float*)alloc(16ull * 512 * 4);
  h16*   Xbuf   = (h16*)  alloc(262144ull * 352 * 2);   // max over SA1/SA2 gathers
  h16*   Xsa3   = (h16*)  alloc(2048ull * 672 * 2);
  h16*   Xg1    = (h16*)  alloc(16ull * 1024 * 2);
  h16*   Xg2    = (h16*)  alloc(16ull * 512 * 2);

  auto pack = [&](const float* Wf, h16* o, int rows, int cin, int cpad) {
    const long total = (long)rows * cpad;
    pack_rows<<<(unsigned)((total + 255) / 256), 256, 0, stream>>>(Wf, o, rows, cin, cpad);
  };
  auto zero = [&](float* p, long n) {
    fill_zero_f32<<<(unsigned)((n + 255) / 256), 256, 0, stream>>>(p, n);
  };

  const int   K1v[3] = {16, 32, 128};
  const float R1v[3] = {0.1f, 0.2f, 0.4f};
  const int   C1[3][3] = {{32, 32, 64}, {64, 64, 128}, {64, 96, 128}};
  const int   K2v[3] = {32, 64, 128};
  const float R2v[3] = {0.2f, 0.4f, 0.8f};
  const int   C2[3][3] = {{64, 64, 128}, {128, 128, 256}, {128, 128, 256}};
  const int   C3[3] = {256, 512, 1024};

  // ---- pack all weights to padded f16 ----
  h16 *w1p[3][3], *w2p[3][3], *w3p[3], *wg1p, *wg2p;
  for (int br = 0; br < 3; ++br)
    for (int l = 0; l < 3; ++l) {
      const int cout = C1[br][l];
      const int cinA = (l == 0) ? 3 : C1[br][l - 1];
      const int cpad = (l == 0) ? 32 : cinA;
      w1p[br][l] = (h16*)alloc((size_t)cout * cpad * 2);
      pack(W1[br][l], w1p[br][l], cout, cinA, cpad);
    }
  for (int br = 0; br < 3; ++br)
    for (int l = 0; l < 3; ++l) {
      const int cout = C2[br][l];
      const int cinA = (l == 0) ? 323 : C2[br][l - 1];
      const int cpad = (l == 0) ? 352 : cinA;
      w2p[br][l] = (h16*)alloc((size_t)cout * cpad * 2);
      pack(W2[br][l], w2p[br][l], cout, cinA, cpad);
    }
  for (int l = 0; l < 3; ++l) {
    const int cout = C3[l];
    const int cinA = (l == 0) ? 643 : C3[l - 1];
    const int cpad = (l == 0) ? 672 : cinA;
    w3p[l] = (h16*)alloc((size_t)cout * cpad * 2);
    pack(W3[l], w3p[l], cout, cinA, cpad);
  }
  wg1p = (h16*)alloc(256ull * 1024 * 2);  pack(g1W, wg1p, 256, 1024, 1024);
  wg2p = (h16*)alloc(256ull * 512 * 2);   pack(g2W, wg2p, 256, 512, 512);

  // allow large dynamic LDS for the fused kernel (up to ~128KB used)
  (void)hipFuncSetAttribute(reinterpret_cast<const void*>(&fused_mlp_pool),
                            hipFuncAttributeMaxDynamicSharedMemorySize, 160 * 1024);

  auto launch_mlp = [&](const h16* X, const h16* Wa, const h16* Wb, const h16* Wc,
                        const float* ga, const float* gb, const float* gc,
                        const float* ba, const float* bb, const float* bc,
                        int d0, int d1, int d2, int d3, int nl,
                        long rows, int groupK, float* pool, int pstride, int poff) {
    MlpArgs a{};
    a.X = X;
    a.W[0] = Wa; a.W[1] = Wb; a.W[2] = Wc;
    a.g[0] = ga; a.g[1] = gb; a.g[2] = gc;
    a.b[0] = ba; a.b[1] = bb; a.b[2] = bc;
    a.cdim[0] = d0; a.cdim[1] = d1; a.cdim[2] = d2; a.cdim[3] = d3;
    a.nlayers = nl; a.rows = rows; a.groupK = groupK;
    a.pool = pool; a.pool_stride = pstride; a.pool_offset = poff;
    int buf0 = d0, buf1 = 0;
    if (nl >= 2) buf1 = d1;
    if (nl >= 3 && d2 > buf0) buf0 = d2;
    a.buf0_elems = buf0;
    const size_t lds = (size_t)64 * (buf0 + buf1) * sizeof(h16);
    const unsigned nblk = (unsigned)((rows + 63) / 64);
    fused_mlp_pool<<<nblk, 128, lds, stream>>>(a);
  };

  // ---- SA1 ----
  fps_kernel<<<16, 256, 0, stream>>>(xyz, 4096, 512, (long)3 * 4096, 1L, 4096L,
                                     idx1, l1_xyz);
  zero(l1_pts, 16L * 512 * 320);
  const int off1[3] = {0, 64, 192};
  for (int br = 0; br < 3; ++br) {
    bq_sa1<<<16 * 512, 32, 0, stream>>>(xyz, l1_xyz, R1v[br] * R1v[br], K1v[br], Xbuf);
    launch_mlp(Xbuf, w1p[br][0], w1p[br][1], w1p[br][2],
               G1[br][0], G1[br][1], G1[br][2], B1[br][0], B1[br][1], B1[br][2],
               32, C1[br][0], C1[br][1], C1[br][2], 3,
               16L * 512 * K1v[br], K1v[br], l1_pts, 320, off1[br]);
  }

  // ---- SA2 ----
  fps_kernel<<<16, 256, 0, stream>>>(l1_xyz, 512, 128, 512L * 3, 3L, 1L,
                                     idx2, l2_xyz);
  zero(l2_pts, 16L * 128 * 640);
  const int off2[3] = {0, 128, 384};
  for (int br = 0; br < 3; ++br) {
    bq_sa2<<<16 * 128, 32, 0, stream>>>(l1_xyz, l1_pts, l2_xyz,
                                        R2v[br] * R2v[br], K2v[br], Xbuf);
    launch_mlp(Xbuf, w2p[br][0], w2p[br][1], w2p[br][2],
               G2[br][0], G2[br][1], G2[br][2], B2[br][0], B2[br][1], B2[br][2],
               352, C2[br][0], C2[br][1], C2[br][2], 3,
               16L * 128 * K2v[br], K2v[br], l2_pts, 640, off2[br]);
  }

  // ---- SA3 (global max over 128 points) ----
  gather_sa3<<<2048, 128, 0, stream>>>(l2_xyz, l2_pts, Xsa3);
  zero(l3, 16L * 1024);
  launch_mlp(Xsa3, w3p[0], w3p[1], w3p[2], G3[0], G3[1], G3[2], B3[0], B3[1], B3[2],
             672, 256, 512, 1024, 3, 2048, 128, l3, 1024, 0);

  // ---- ghost1 ----
  pack(l3, Xg1, 16, 1024, 1024);
  zero(x1a, 16L * 256);
  launch_mlp(Xg1, wg1p, nullptr, nullptr, g1g, nullptr, nullptr, g1b, nullptr, nullptr,
             1024, 256, 0, 0, 1, 16, 1, x1a, 256, 0);
  ghost_cheap<<<16, 256, 0, stream>>>(x1a, g1Wc, g1gc, g1bc, f1);

  // ---- ghost2 ----
  pack(f1, Xg2, 16, 512, 512);
  zero(x1b, 16L * 256);
  launch_mlp(Xg2, wg2p, nullptr, nullptr, g2g, nullptr, nullptr, g2b, nullptr, nullptr,
             512, 256, 0, 0, 1, 16, 1, x1b, 256, 0);
  ghost_cheap<<<16, 256, 0, stream>>>(x1b, g2Wc, g2gc, g2bc, f2);

  // ---- fc3 + log_softmax + output assembly: [logits, l3, f] ----
  float* outp = (float*)d_out;
  fc3_logsoftmax<<<16, 64, 0, stream>>>(f2, fc3w, fc3b, outp);
  copy_f32<<<(16 * 1024 + 255) / 256, 256, 0, stream>>>(l3, outp + 640, 16L * 1024);
  copy_f32<<<(16 * 512 + 255) / 256, 256, 0, stream>>>(f2, outp + 640 + 16 * 1024, 16L * 512);
}
